// VectorQuantizer_56246891708999
// MI455X (gfx1250) — matrix operand order repaired and checked by reading, not compiled, not run on hardware
//
#include <hip/hip_runtime.h>
#include <hip/hip_bf16.h>

// ---------------------------------------------------------------------------
// VQ-VAE vector quantizer for gfx1250 (MI455X).
//   z_e:       [32,32,32,256] f32  -> N=32768 rows, D=256
//   embedding: [4096,256]     f32  -> K=4096 codes
// Outputs (concatenated in d_out, f32): z_q [N*256], indices [N], perplexity.
//
// Distance matmul runs on v_wmma_f32_16x16x32_f16 with compensated split-f16
// operands (hi+lo residual), ~f32 accuracy at f16-WMMA rate.
// Codebook is pre-split to f16 hi/lo in workspace; the main loop stages B
// tiles into LDS with the Tensor Data Mover (tensor_load_to_lds, TENSORcnt),
// double buffered. sched_group_barrier pipelines DS reads ahead of WMMAs and
// dual accumulators break the WMMA dependency chain.
// ---------------------------------------------------------------------------

typedef _Float16 v16h __attribute__((ext_vector_type(16)));
typedef float    v8f  __attribute__((ext_vector_type(8)));
typedef unsigned int u32x4 __attribute__((ext_vector_type(4)));
typedef int          i32x8 __attribute__((ext_vector_type(8)));
typedef int          i32x4 __attribute__((ext_vector_type(4)));

union V16H { uint4 q[2]; v16h h; };

#define VQ_N       32768
#define VQ_K       4096
#define VQ_D       256
#define COL_TILES  256          // K / 16
#define LDS_ROW_U  132          // uints per staged B row: 128 data + 4 pad

__device__ __forceinline__ unsigned pack2(_Float16 a, _Float16 b) {
  union { _Float16 h[2]; unsigned u; } x;
  x.h[0] = a; x.h[1] = b;
  return x.u;
}

// LDS byte offset of a __shared__ object: AMDGPU local->flat addrspacecast
// places the LDS offset in the low 32 bits of the generic pointer.
__device__ __forceinline__ unsigned lds_off_of(const void* p) {
  return (unsigned)(uintptr_t)p;
}

// TDM: DMA one 256x16 f16 tile (row stride 256 elems) from global to LDS,
// padding 4 DWORDs after every 128 DWORDs (512B row) in LDS.
__device__ __forceinline__ void tdm_load_tile(unsigned lds_byte_off,
                                              const _Float16* gsrc) {
  unsigned long long ga = (unsigned long long)(uintptr_t)gsrc;
  u32x4 g0 = { 1u,                                  // count=1, user mode
               lds_byte_off,                        // lds_addr
               (unsigned)ga,                        // global_addr[31:0]
               ((unsigned)(ga >> 32) & 0x01FFFFFFu) // global_addr[56:32]
                   | 0x80000000u };                 // type=2 ("image")
  i32x8 g1 = { 0x07910000,   // data_size=1(2B), pad_en, pad_intv=6, pad_amt=3
               0x01000000,   // tensor_dim0 = 256 (bits 79:48 low half)
               0x00100000,   // tensor_dim1 = 16  (bits 111:80 low half)
               0x01000000,   // tile_dim0   = 256 (bits 127:112)
               0x00000010,   // tile_dim1   = 16, tile_dim2 = 0
               256,          // tensor_dim0_stride (48b, low dword)
               0, 0 };
  i32x4 gz4 = { 0, 0, 0, 0 };     // groups 2/3 unused (2-D tile)
  i32x8 gz8 = { 0, 0, 0, 0, 0, 0, 0, 0 };
  __builtin_amdgcn_tensor_load_to_lds(g0, g1, gz4, gz4, gz8, 0);
}

// ---------------------------------------------------------------------------
__global__ void vq_zero(int* __restrict__ counts) {
  counts[blockIdx.x * 256 + threadIdx.x] = 0;
}

// Split codebook to f16 hi/lo and compute ||e_k||^2. One wave per code row.
__global__ void vq_prep(const float* __restrict__ emb,
                        _Float16* __restrict__ hi,
                        _Float16* __restrict__ lo,
                        float* __restrict__ enorm) {
  const int wave = threadIdx.x >> 5;
  const int lane = threadIdx.x & 31;
  const int row  = blockIdx.x * 8 + wave;
  const float* p = emb + (size_t)row * VQ_D + lane * 8;
  float4 a = ((const float4*)p)[0];
  float4 b = ((const float4*)p)[1];
  float f[8] = { a.x, a.y, a.z, a.w, b.x, b.y, b.z, b.w };
  unsigned hw[4], lw[4];
  float s = 0.f;
#pragma unroll
  for (int j = 0; j < 4; ++j) {
    float x0 = f[2 * j], x1 = f[2 * j + 1];
    s = fmaf(x0, x0, s);
    s = fmaf(x1, x1, s);
    _Float16 h0 = (_Float16)x0, h1 = (_Float16)x1;
    hw[j] = pack2(h0, h1);
    lw[j] = pack2((_Float16)(x0 - (float)h0), (_Float16)(x1 - (float)h1));
  }
  *(uint4*)(hi + (size_t)row * VQ_D + lane * 8) =
      make_uint4(hw[0], hw[1], hw[2], hw[3]);
  *(uint4*)(lo + (size_t)row * VQ_D + lane * 8) =
      make_uint4(lw[0], lw[1], lw[2], lw[3]);
#pragma unroll
  for (int off = 1; off < 32; off <<= 1) s += __shfl_xor(s, off, 32);
  if (lane == 0) enorm[row] = s;
}

// ---------------------------------------------------------------------------
// Main kernel: 256 blocks x 256 threads (8 waves). Wave w owns rows
// [blk*128 + w*16, +16) and sweeps all 256 codebook column tiles.
// ---------------------------------------------------------------------------
__global__ __launch_bounds__(256)
void vq_main(const float* __restrict__ z,
             const _Float16* __restrict__ embHi,
             const _Float16* __restrict__ embLo,
             const float* __restrict__ enorm,
             float* __restrict__ idx_out,
             int* __restrict__ counts) {
  __shared__ unsigned sHi[2][16 * LDS_ROW_U];
  __shared__ unsigned sLo[2][16 * LDS_ROW_U];

  const int tid  = threadIdx.x;
  const int wave = tid >> 5;
  const int lane = tid & 31;
  const int lmod = lane & 15;
  const int half = lane >> 4;
  const int rowBase = blockIdx.x * 128 + wave * 16;

  // ---- A fragments: z rows -> registers as split f16 (ISA 16-bit A layout:
  // lane L holds row M=L%16; VGPR v holds K = (v<4 ? 2v : 16+2(v-4)) + 8*(L/16))
  v16h a_hi[8], a_lo[8];
  float psum = 0.f;
  const float* zrow = z + (size_t)(rowBase + lmod) * VQ_D;
#pragma unroll
  for (int c = 0; c < 8; ++c) {
    const int k0 = 32 * c + 8 * half;
    float4 f0a = *(const float4*)(zrow + k0);
    float4 f0b = *(const float4*)(zrow + k0 + 4);
    float4 f1a = *(const float4*)(zrow + k0 + 16);
    float4 f1b = *(const float4*)(zrow + k0 + 20);
    float f[16] = { f0a.x, f0a.y, f0a.z, f0a.w,  f0b.x, f0b.y, f0b.z, f0b.w,
                    f1a.x, f1a.y, f1a.z, f1a.w,  f1b.x, f1b.y, f1b.z, f1b.w };
#pragma unroll
    for (int j = 0; j < 16; ++j) {
      psum = fmaf(f[j], f[j], psum);
      _Float16 h = (_Float16)f[j];
      a_hi[c][j] = h;
      a_lo[c][j] = (_Float16)(f[j] - (float)h);
    }
  }
  // ||z_row||^2 : row M=L%16 is split across lanes L and L^16.
  float fullnorm = psum + __shfl_xor(psum, 16, 32);
  float znr[8];
#pragma unroll
  for (int r = 0; r < 8; ++r)
    znr[r] = __shfl(fullnorm, r + 8 * half, 32);   // norm of C row M=r+8*half

  float minv[8];
  int   mini[8];
#pragma unroll
  for (int r = 0; r < 8; ++r) { minv[r] = 3.4e38f; mini[r] = 0; }

  // ---- preload tile 0 via TDM (wave 0 only; EXEC is ignored by TDM but the
  // instruction is per-wave, so gate by wave id to avoid 8x duplicate DMA).
  if (wave == 0) {
    tdm_load_tile(lds_off_of(&sHi[0][0]), embHi);
    tdm_load_tile(lds_off_of(&sLo[0][0]), embLo);
    __builtin_amdgcn_s_wait_tensorcnt(0);
  }
  __syncthreads();

  for (int t = 0; t < COL_TILES; ++t) {
    const int buf = t & 1;
    // kick DMA for tile t+1 into the other buffer; overlaps with compute
    if (wave == 0 && t + 1 < COL_TILES) {
      const size_t off = (size_t)(t + 1) * 16 * VQ_D;
      tdm_load_tile(lds_off_of(&sHi[buf ^ 1][0]), embHi + off);
      tdm_load_tile(lds_off_of(&sLo[buf ^ 1][0]), embLo + off);
    }

    const float en = enorm[t * 16 + lmod];

    // B layout (32x16, 16-bit): lane L holds code row N=L%16; VGPR v holds
    // K = (v<4 ? 2v : 16+2(v-4)) + 8*(L/16) in-chunk (matches A).
    const unsigned* hbase = &sHi[buf][lmod * LDS_ROW_U + 4 * half];
    const unsigned* lbase = &sLo[buf][lmod * LDS_ROW_U + 4 * half];

    // Dual accumulators: acc_h carries zh*eh, acc_l carries zl*eh + zh*el,
    // so consecutive WMMAs alternate dependency chains.
    v8f acc_h = {}, acc_l = {};
#pragma unroll
    for (int c = 0; c < 8; ++c) {
      V16H bh, bl;
      bh.q[0] = *(const uint4*)(hbase + 16 * c);
      bh.q[1] = *(const uint4*)(hbase + 16 * c + 8);
      bl.q[0] = *(const uint4*)(lbase + 16 * c);
      bl.q[1] = *(const uint4*)(lbase + 16 * c + 8);
      acc_h = __builtin_amdgcn_wmma_f32_16x16x32_f16(false, a_hi[c], false,
                                                     bh.h, (short)0, acc_h,
                                                     false, false);
      acc_l = __builtin_amdgcn_wmma_f32_16x16x32_f16(false, a_lo[c], false,
                                                     bh.h, (short)0, acc_l,
                                                     false, false);
      acc_l = __builtin_amdgcn_wmma_f32_16x16x32_f16(false, a_hi[c], false,
                                                     bl.h, (short)0, acc_l,
                                                     false, false);
    }
    // Scheduler pipeline: DS4, (DS4, WMMA3)x7, WMMA3 -> one chunk of DS-read
    // lookahead so s_wait_dscnt overlaps WMMA issue instead of stalling.
    __builtin_amdgcn_sched_group_barrier(0x0100, 4, 0);   // DS read x4
#pragma unroll
    for (int c = 0; c < 7; ++c) {
      __builtin_amdgcn_sched_group_barrier(0x0100, 4, 0); // DS read x4
      __builtin_amdgcn_sched_group_barrier(0x0008, 3, 0); // WMMA x3
    }
    __builtin_amdgcn_sched_group_barrier(0x0008, 3, 0);   // WMMA x3

    // C/D layout: acc[r] is (M = r + 8*half, N = lane%16).
    const int nglob = t * 16 + lmod;
#pragma unroll
    for (int r = 0; r < 8; ++r) {
      float d = fmaf(-2.0f, acc_h[r] + acc_l[r], znr[r] + en);
      if (d < minv[r]) { minv[r] = d; mini[r] = nglob; }
    }

    if (wave == 0) __builtin_amdgcn_s_wait_tensorcnt(0); // tile t+1 landed
    __syncthreads();
  }

  // ---- argmin reduce across the 16 lanes holding row M=r+8*half
#pragma unroll
  for (int r = 0; r < 8; ++r) {
    float v = minv[r];
    int   i = mini[r];
#pragma unroll
    for (int off = 1; off < 16; off <<= 1) {
      float ov = __shfl_xor(v, off, 32);
      int   oi = __shfl_xor(i, off, 32);
      if (ov < v || (ov == v && oi < i)) { v = ov; i = oi; }
    }
    minv[r] = v; mini[r] = i;
  }
  if (lmod == 0) {
#pragma unroll
    for (int r = 0; r < 8; ++r) {
      const int row = rowBase + r + 8 * half;
      idx_out[row] = (float)mini[r];
      atomicAdd(&counts[mini[r]], 1);
    }
  }
}

// ---------------------------------------------------------------------------
__global__ void vq_gather(const float* __restrict__ emb,
                          const float* __restrict__ idxF,
                          float* __restrict__ zq) {
  const int row = blockIdx.x * 4 + (threadIdx.x >> 6);
  const int col = (threadIdx.x & 63) * 4;
  const int k   = (int)idxF[row];
  *(float4*)(zq + (size_t)row * VQ_D + col) =
      *(const float4*)(emb + (size_t)k * VQ_D + col);
}

__global__ void vq_perp(const int* __restrict__ counts,
                        float* __restrict__ out) {
  __shared__ float red[256];
  float s = 0.f;
  for (int b = threadIdx.x; b < VQ_K; b += 256) {
    float p = (float)counts[b] * (1.0f / (float)VQ_N);
    s += p * logf(p + 1e-10f);
  }
  red[threadIdx.x] = s;
  __syncthreads();
  for (int st = 128; st > 0; st >>= 1) {
    if (threadIdx.x < st) red[threadIdx.x] += red[threadIdx.x + st];
    __syncthreads();
  }
  if (threadIdx.x == 0) out[0] = expf(-red[0]);
}

// ---------------------------------------------------------------------------
extern "C" void kernel_launch(void* const* d_in, const int* in_sizes, int n_in,
                              void* d_out, int out_size, void* d_ws,
                              size_t ws_size, hipStream_t stream) {
  const float* z   = (const float*)d_in[0];   // [32768,256]
  const float* emb = (const float*)d_in[1];   // [4096,256]

  float* zq   = (float*)d_out;                // [32768*256]
  float* idxF = zq + (size_t)VQ_N * VQ_D;     // [32768]
  float* perp = idxF + VQ_N;                  // [1]

  // workspace: embHi (2MB) | embLo (2MB) | enorm (16KB) | counts (16KB)
  _Float16* embHi = (_Float16*)d_ws;
  _Float16* embLo = embHi + (size_t)VQ_K * VQ_D;
  float*    enorm = (float*)(embLo + (size_t)VQ_K * VQ_D);
  int*      counts = (int*)(enorm + VQ_K);

  vq_zero  <<<VQ_K / 256, 256, 0, stream>>>(counts);
  vq_prep  <<<VQ_K / 8,   256, 0, stream>>>(emb, embHi, embLo, enorm);
  vq_main  <<<VQ_N / 128, 256, 0, stream>>>(z, embHi, embLo, enorm, idxF, counts);
  vq_gather<<<VQ_N / 4,   256, 0, stream>>>(emb, idxF, zq);
  vq_perp  <<<1,          256, 0, stream>>>(counts, perp);
}
